// TRPN_81166291960188
// MI455X (gfx1250) — matrix operand order repaired and checked by reading, not compiled, not run on hardware
//
#include <hip/hip_runtime.h>
#include <hip/hip_bf16.h>

// ---------------------------------------------------------------------------
// TRPN forward for MI455X (gfx1250, wave32, WMMA bf16 16x16x32, f32 accum)
// ---------------------------------------------------------------------------

typedef __attribute__((ext_vector_type(16))) __bf16        v16bf;
typedef __attribute__((ext_vector_type(8)))  float         v8f;
typedef __attribute__((ext_vector_type(8)))  unsigned int  v8u;
typedef __attribute__((ext_vector_type(4)))  unsigned int  v4u;
typedef __attribute__((ext_vector_type(8)))  int           v8i;
typedef __attribute__((ext_vector_type(4)))  int           v4i;

#define WMMA_BF16(a, b, c) \
  __builtin_amdgcn_wmma_f32_16x16x32_bf16(false, (a), false, (b), (short)0, (c), false, false)

__device__ __forceinline__ unsigned short f2bf(float f) {
  unsigned u = __float_as_uint(f);
  u += 0x7FFFu + ((u >> 16) & 1u);          // round-to-nearest-even
  return (unsigned short)(u >> 16);
}
__device__ __forceinline__ float bf2f(unsigned short b) {
  return __uint_as_float(((unsigned)b) << 16);
}

// Packed B fragment (32x16 bf16): 16 contiguous bf16 (32B) per lane.
__device__ __forceinline__ v16bf load_b(const unsigned short* p) {
  v8u r = *(const v8u*)p;
  return __builtin_bit_cast(v16bf, r);
}

// Gather A fragment (16x32 bf16) from a bf16 row (this lane's row m = lane&15).
// Interleaved 16-bit A layout: VGPR v holds K = 16*(v>=4) + 8*half + 2*(v&3) (+0,+1).
__device__ __forceinline__ v16bf gather_a(const unsigned short* rowbase, int kbase, int half) {
  v8u r;
#pragma unroll
  for (int v = 0; v < 8; ++v) {
    int k = kbase + ((v & 4) << 2) + 8 * half + 2 * (v & 3);
    r[v] = *(const unsigned int*)(rowbase + k);
  }
  return __builtin_bit_cast(v16bf, r);
}

// ---------------------------------------------------------------------------
// Workspace layout (bytes). All fragment bases are 1024B aligned -> every
// per-lane 32B fragment load is 32B aligned.
// ---------------------------------------------------------------------------
constexpr size_t OFF_W11 = 0;                                  // 8*40*512  bf16
constexpr size_t SZ_W11  = 163840ull * 2;
constexpr size_t OFF_W12 = OFF_W11 + SZ_W11;                   // 20*20*512 bf16
constexpr size_t SZ_W12  = 204800ull * 2;
constexpr size_t OFF_WG  = OFF_W12 + SZ_W12;                   // 124*248*512 bf16
constexpr size_t SZ_WG   = 15745024ull * 2;
constexpr size_t OFF_W21 = OFF_WG + SZ_WG;                     // 124*20*512 bf16
constexpr size_t SZ_W21  = 1269760ull * 2;
constexpr size_t OFF_W22 = OFF_W21 + SZ_W21;                   // 10*10*512 bf16
constexpr size_t SZ_W22  = 51200ull * 2;
constexpr size_t OFF_W23 = OFF_W22 + SZ_W22;                   // 5*2*512 bf16 (cols padded to 32)
constexpr size_t SZ_W23  = 5120ull * 2;
constexpr size_t OFF_QT  = ((OFF_W23 + SZ_W23 + 255) / 256) * 256;  // qterm [64,3968] f32
constexpr size_t SZ_QT   = 64ull * 3968 * 4;
constexpr size_t OFF_SUP = OFF_QT + SZ_QT;                     // support, B-frag packed, bf16
constexpr size_t SZ_SUP  = 64ull * 496 * 512 * 2;              // per task: 248 nt * 2 kt * 512
constexpr size_t OFF_FB  = OFF_SUP + SZ_SUP;                   // node_feat bf16 [64*60*128]
constexpr size_t SZ_FB   = 491520ull * 2;

// ---------------------------------------------------------------------------
// Pack a row-major f32 weight [Ksrc x Nsrc] into WMMA B-fragment layout.
// ---------------------------------------------------------------------------
__global__ void k_pack(const float* __restrict__ src, unsigned short* __restrict__ dst,
                       int Ksrc, int Nsrc, int NT, int total) {
  int gid = blockIdx.x * 256 + threadIdx.x;
  if (gid >= total) return;
  int e    = gid & 15;
  int lane = (gid >> 4) & 31;
  int f    = gid >> 9;
  int kt   = f / NT, nt = f % NT;
  int k    = kt * 32 + ((lane >> 4) << 3) + (e & 7) + ((e >> 3) << 4);
  int col  = nt * 16 + (lane & 15);
  float v  = (k < Ksrc && col < Nsrc) ? src[(size_t)k * Nsrc + col] : 0.f;
  dst[gid] = f2bf(v);
}

// Plain f32 -> bf16 pack (node_feat).
__global__ void k_packf(const float* __restrict__ src, unsigned short* __restrict__ dst, int total) {
  int gid = blockIdx.x * 256 + threadIdx.x;
  if (gid < total) dst[gid] = f2bf(src[gid]);
}

// ---------------------------------------------------------------------------
// Pairwise MLP -> learned_score [T,N,N] at out[115200 + r].
// One wave per 32 rows (two 16-row A tiles share every B fragment -> halves
// the L2 B-fragment traffic, which is the binding resource on this kernel).
// ---------------------------------------------------------------------------
__global__ __launch_bounds__(32) void k_pair(const unsigned short* __restrict__ featb,
    const float* __restrict__ W13, const float* __restrict__ b11,
    const float* __restrict__ b12, const float* __restrict__ b13,
    const unsigned short* __restrict__ pW11, const unsigned short* __restrict__ pW12,
    float* __restrict__ out) {
  extern __shared__ unsigned short smem[];
  unsigned short* xs = smem;            // 32*256 = 8192
  unsigned short* h1 = xs + 8192;       // 32*640 = 20480
  unsigned short* h2 = h1 + 20480;      // 32*320 = 10240   (total 77824 B)
  const int lane = threadIdx.x;
  const int R0   = blockIdx.x * 32;
  const int half = lane >> 4, m = lane & 15, c = lane & 15;

  // stage 32 rows of [feat_i | feat_j] as raw bf16 dwords (no conversion)
  const unsigned* fb32 = (const unsigned*)featb;
  unsigned* xs32 = (unsigned*)xs;
  for (int idx = lane; idx < 32 * 128; idx += 32) {
    int mm = idx >> 7, cp = idx & 127;
    int r = R0 + mm, t = r / 3600, rem = r % 3600, i = rem / 60, j = rem % 60;
    int sel = (cp < 64) ? i : j;
    xs32[idx] = fb32[(size_t)(t * 60 + sel) * 64 + (cp & 63)];
  }
  __syncthreads();

  v16bf af[2][8];
#pragma unroll
  for (int mt = 0; mt < 2; ++mt)
#pragma unroll
    for (int kt = 0; kt < 8; ++kt)
      af[mt][kt] = gather_a(&xs[(mt * 16 + m) * 256], kt * 32, half);

  // layer 1: K=256, Nout=640
  for (int nt = 0; nt < 40; ++nt) {
    v8f acc0 = {}, acc1 = {};
#pragma unroll
    for (int kt = 0; kt < 8; ++kt) {
      v16bf b = load_b(pW11 + (size_t)(kt * 40 + nt) * 512 + lane * 16);
      acc0 = WMMA_BF16(af[0][kt], b, acc0);
      acc1 = WMMA_BF16(af[1][kt], b, acc1);
    }
    float bias = b11[nt * 16 + c];
#pragma unroll
    for (int v = 0; v < 8; ++v) {
      float h = acc0[v] + bias; h = h > 0.f ? h : 0.f;
      h1[(v + 8 * half) * 640 + nt * 16 + c] = f2bf(h);
      h = acc1[v] + bias; h = h > 0.f ? h : 0.f;
      h1[(16 + v + 8 * half) * 640 + nt * 16 + c] = f2bf(h);
    }
  }
  __syncthreads();

  // layer 2: K=640, Nout=320
  for (int nt = 0; nt < 20; ++nt) {
    v8f acc0 = {}, acc1 = {};
    for (int kt = 0; kt < 20; ++kt) {
      v16bf b = load_b(pW12 + (size_t)(kt * 20 + nt) * 512 + lane * 16);
      acc0 = WMMA_BF16(gather_a(&h1[m * 640], kt * 32, half), b, acc0);
      acc1 = WMMA_BF16(gather_a(&h1[(16 + m) * 640], kt * 32, half), b, acc1);
    }
    float bias = b12[nt * 16 + c];
#pragma unroll
    for (int v = 0; v < 8; ++v) {
      float h = acc0[v] + bias; h = h > 0.f ? h : 0.f;
      h2[(v + 8 * half) * 320 + nt * 16 + c] = f2bf(h);
      h = acc1[v] + bias; h = h > 0.f ? h : 0.f;
      h2[(16 + v + 8 * half) * 320 + nt * 16 + c] = f2bf(h);
    }
  }
  __syncthreads();

  // layer 3: 320 -> 1, two 16-row groups, halves combined via wave shuffle
#pragma unroll
  for (int g = 0; g < 2; ++g) {
    float partial = 0.f;
    const unsigned short* hr = &h2[(g * 16 + m) * 320 + half * 160];
    for (int k = 0; k < 160; ++k) partial += bf2f(hr[k]) * W13[half * 160 + k];
    float tot = partial + __shfl_xor(partial, 16, 32);
    if (half == 0) {
      float s = 1.f / (1.f + __expf(-(tot + b13[0])));
      out[115200 + R0 + g * 16 + m] = s;
    }
  }
}

// ---------------------------------------------------------------------------
// qterm[t,:] = flatten(feat[t,30:60,:]) @ Wg[128:3968,:]   (M=64, K=3840, N=3968)
// A rows are contiguous bf16 slices of featb -> raw dword gathers, no cvt.
// ---------------------------------------------------------------------------
__global__ __launch_bounds__(32) void k_qterm(const unsigned short* __restrict__ featb,
    const unsigned short* __restrict__ pWg, float* __restrict__ qterm) {
  const int lane = threadIdx.x, half = lane >> 4, m = lane & 15, c = lane & 15;
  const int nt = blockIdx.x, mt = blockIdx.y;
  const unsigned* arow32 = (const unsigned*)(featb + (size_t)(mt * 16 + m) * 7680 + 3840);
  v8f acc = {};
  for (int kt = 0; kt < 120; ++kt) {
    v8u r;
#pragma unroll
    for (int v = 0; v < 8; ++v) {
      int k = kt * 32 + ((v & 4) << 2) + 8 * half + 2 * (v & 3);
      r[v] = arow32[k >> 1];
    }
    v16bf a = __builtin_bit_cast(v16bf, r);
    acc = WMMA_BF16(a, load_b(pWg + (size_t)((kt + 4) * 248 + nt) * 512 + lane * 16), acc);
  }
#pragma unroll
  for (int v = 0; v < 8; ++v)
    qterm[(size_t)(mt * 16 + v + 8 * half) * 3968 + nt * 16 + c] = acc[v];
}

// ---------------------------------------------------------------------------
// support[t,m,g] = feat[t,m,:]@Wg[0:128,g] + qterm[t,g] + bg[g]; stored bf16
// in B-fragment layout. The 60x128 bf16 feature tile (15360 contiguous bytes)
// is staged into LDS by the Tensor Data Mover (1-D D# descriptor).
// ---------------------------------------------------------------------------
__global__ __launch_bounds__(32) void k_support(const unsigned short* __restrict__ featb,
    const unsigned short* __restrict__ pWg, const float* __restrict__ qterm,
    const float* __restrict__ bg, unsigned short* __restrict__ supf) {
  __shared__ unsigned short fs[64 * 128];
  __shared__ float sup[64 * 16];
  const int lane = threadIdx.x, half = lane >> 4, mm = lane & 15, c = lane & 15;
  const int nt = blockIdx.x, t = blockIdx.y;

  // zero the 4 pad rows (disjoint from the TDM destination range)
  for (int idx = lane; idx < 4 * 128; idx += 32) fs[60 * 128 + idx] = 0;

#if __has_builtin(__builtin_amdgcn_tensor_load_to_lds)
  {
    // D# group0: count=1 | lds_addr | global_addr | type=2
    unsigned lds_off = (unsigned)(unsigned long long)(void*)&fs[0];
    unsigned long long ga = (unsigned long long)(const void*)(featb + (size_t)t * 7680);
    v4u g0;
    g0[0] = 1u;                                   // count=1, no gather
    g0[1] = lds_off;                              // lds_addr
    g0[2] = (unsigned)ga;                         // global_addr[31:0]
    g0[3] = (unsigned)(ga >> 32) | (2u << 30);    // global_addr[56:32] | type=2
    // D# group1: data_size=2B, tensor_dim0=tile_dim0=7680 elements, 1-D tile
    v8i g1;
    g1[0] = 0x00010000;                           // wg_mask=0, data_size=1 (2B)
    g1[1] = (int)(7680u << 16);                   // tensor_dim0[15:0] << 16
    g1[2] = (int)((7680u >> 16) | (1u << 16));    // tensor_dim0[31:16] | tensor_dim1=1
    g1[3] = (int)(7680u << 16);                   // tile_dim0 = 7680
    g1[4] = 0;                                    // tile_dim1/2 unused
    g1[5] = 7680;                                 // tensor_dim0_stride
    g1[6] = 0;
    g1[7] = 0;
    v4i z4 = {0, 0, 0, 0};
#if __clang_major__ >= 23
    v8i z8 = {0, 0, 0, 0, 0, 0, 0, 0};
    __builtin_amdgcn_tensor_load_to_lds(g0, g1, z4, z4, z8, 0);
#else
    __builtin_amdgcn_tensor_load_to_lds(g0, g1, z4, z4, 0);
#endif
    __builtin_amdgcn_s_wait_tensorcnt(0);
  }
#else
  for (int idx = lane; idx < 60 * 64; idx += 32)
    ((unsigned*)fs)[idx] = ((const unsigned*)(featb + (size_t)t * 7680))[idx];
#endif
  __syncthreads();

  float addv = qterm[(size_t)t * 3968 + nt * 16 + c] + bg[nt * 16 + c];
#pragma unroll
  for (int mt = 0; mt < 4; ++mt) {
    v8f acc = {};
#pragma unroll
    for (int kt = 0; kt < 4; ++kt) {
      v16bf a = gather_a(&fs[(mt * 16 + mm) * 128], kt * 32, half);
      acc = WMMA_BF16(a, load_b(pWg + (size_t)(kt * 248 + nt) * 512 + lane * 16), acc);
    }
#pragma unroll
    for (int v = 0; v < 8; ++v)
      sup[(mt * 16 + v + 8 * half) * 16 + c] = acc[v] + addv;
  }
  __syncthreads();

  unsigned short* dst = supf + (size_t)(t * 496 + nt * 2) * 512;
#pragma unroll
  for (int kt = 0; kt < 2; ++kt) {
#pragma unroll
    for (int e = 0; e < 16; ++e) {
      int mr = kt * 32 + 8 * half + (e & 7) + ((e >> 3) << 4);
      dst[kt * 512 + lane * 16 + e] = (mr < 60) ? f2bf(sup[mr * 16 + c]) : (unsigned short)0;
    }
  }
}

// ---------------------------------------------------------------------------
// Fused tail per (t, mt): learned_adj rows, adjacency GEMM (K=64), relu,
// fc_2 (3968->320->160->30), sigmoid, transposed store.
// h2 row-tile (16x3968 bf16) lives entirely in dynamic LDS (~144KB < 320KB).
// ---------------------------------------------------------------------------
__global__ __launch_bounds__(32) void k_tail(const float* __restrict__ adj,
    const float* __restrict__ score, const unsigned short* __restrict__ supf,
    const unsigned short* __restrict__ pW21, const unsigned short* __restrict__ pW22,
    const unsigned short* __restrict__ pW23, const float* __restrict__ b21,
    const float* __restrict__ b22, const float* __restrict__ b23,
    float* __restrict__ out) {
  extern __shared__ unsigned short smem[];
  unsigned short* adjA = smem;            // 16*64   = 1024
  unsigned short* h2   = adjA + 1024;     // 16*3968 = 63488
  unsigned short* h3   = h2 + 63488;      // 16*320  = 5120
  unsigned short* h4   = h3 + 5120;       // 16*160  = 2560
  const int lane = threadIdx.x, half = lane >> 4, m = lane & 15, c = lane & 15;
  const int mt = blockIdx.x, t = blockIdx.y;

  for (int idx = lane; idx < 16 * 64; idx += 32) {
    int il = idx >> 6, mm = idx & 63;
    int i = mt * 16 + il;
    float v = 0.f;
    if (i < 60 && mm < 60) {
      float sc = score[(size_t)t * 3600 + i * 60 + mm];
      if (i < 30 && mm < 30) v = (adj[(size_t)t * 900 + i * 30 + mm] > 0.f) ? 1.f : -sc;
      else                   v = sc;
    }
    adjA[idx] = f2bf(v);
  }
  __syncthreads();

  v16bf af0 = gather_a(&adjA[m * 64], 0, half);
  v16bf af1 = gather_a(&adjA[m * 64], 32, half);
  const unsigned short* sf = supf + (size_t)t * 496 * 512;

  for (int nt = 0; nt < 248; ++nt) {
    v8f acc = {};
    acc = WMMA_BF16(af0, load_b(sf + (size_t)(nt * 2 + 0) * 512 + lane * 16), acc);
    acc = WMMA_BF16(af1, load_b(sf + (size_t)(nt * 2 + 1) * 512 + lane * 16), acc);
#pragma unroll
    for (int v = 0; v < 8; ++v) {
      float h = acc[v]; h = h > 0.f ? h : 0.f;                 // relu(gc_out)
      h2[(v + 8 * half) * 3968 + nt * 16 + c] = f2bf(h);
    }
  }
  __syncthreads();

  // fc2 layer 1: K=3968, Nout=320
  for (int nt = 0; nt < 20; ++nt) {
    v8f acc = {};
    for (int kt = 0; kt < 124; ++kt) {
      v16bf a = gather_a(&h2[m * 3968], kt * 32, half);
      acc = WMMA_BF16(a, load_b(pW21 + (size_t)(kt * 20 + nt) * 512 + lane * 16), acc);
    }
    float bias = b21[nt * 16 + c];
#pragma unroll
    for (int v = 0; v < 8; ++v) {
      float h = acc[v] + bias; h = h > 0.f ? h : 0.f;
      h3[(v + 8 * half) * 320 + nt * 16 + c] = f2bf(h);
    }
  }
  __syncthreads();

  // fc2 layer 2: K=320, Nout=160
  for (int nt = 0; nt < 10; ++nt) {
    v8f acc = {};
#pragma unroll
    for (int kt = 0; kt < 10; ++kt) {
      v16bf a = gather_a(&h3[m * 320], kt * 32, half);
      acc = WMMA_BF16(a, load_b(pW22 + (size_t)(kt * 10 + nt) * 512 + lane * 16), acc);
    }
    float bias = b22[nt * 16 + c];
#pragma unroll
    for (int v = 0; v < 8; ++v) {
      float h = acc[v] + bias; h = h > 0.f ? h : 0.f;
      h4[(v + 8 * half) * 160 + nt * 16 + c] = f2bf(h);
    }
  }
  __syncthreads();

  // fc2 layer 3: K=160, Nout=30 (padded 32), sigmoid, transposed store
#pragma unroll
  for (int nt = 0; nt < 2; ++nt) {
    v8f acc = {};
#pragma unroll
    for (int kt = 0; kt < 5; ++kt) {
      v16bf a = gather_a(&h4[m * 160], kt * 32, half);
      acc = WMMA_BF16(a, load_b(pW23 + (size_t)(kt * 2 + nt) * 512 + lane * 16), acc);
    }
    int q = nt * 16 + c;
    if (q < 30) {
      float bias = b23[q];
#pragma unroll
      for (int v = 0; v < 8; ++v) {
        int n = mt * 16 + v + 8 * half;
        if (n < 60) {
          float s = 1.f / (1.f + __expf(-(acc[v] + bias)));
          out[(size_t)t * 1800 + q * 60 + n] = s;     // query_score[t,q,n]
        }
      }
    }
  }
}

// ---------------------------------------------------------------------------
extern "C" void kernel_launch(void* const* d_in, const int* in_sizes, int n_in,
                              void* d_out, int out_size, void* d_ws, size_t ws_size,
                              hipStream_t stream) {
  (void)in_sizes; (void)n_in; (void)out_size; (void)ws_size;
  const float* feat = (const float*)d_in[0];
  const float* adj  = (const float*)d_in[1];
  const float* W11  = (const float*)d_in[2];
  const float* b11  = (const float*)d_in[3];
  const float* W12  = (const float*)d_in[4];
  const float* b12  = (const float*)d_in[5];
  const float* W13  = (const float*)d_in[6];
  const float* b13  = (const float*)d_in[7];
  const float* Wg   = (const float*)d_in[8];
  const float* bg   = (const float*)d_in[9];
  const float* W21  = (const float*)d_in[10];
  const float* b21  = (const float*)d_in[11];
  const float* W22  = (const float*)d_in[12];
  const float* b22  = (const float*)d_in[13];
  const float* W23  = (const float*)d_in[14];
  const float* b23  = (const float*)d_in[15];
  float* out = (float*)d_out;
  char*  ws  = (char*)d_ws;

  unsigned short* pW11 = (unsigned short*)(ws + OFF_W11);
  unsigned short* pW12 = (unsigned short*)(ws + OFF_W12);
  unsigned short* pWg  = (unsigned short*)(ws + OFF_WG);
  unsigned short* pW21 = (unsigned short*)(ws + OFF_W21);
  unsigned short* pW22 = (unsigned short*)(ws + OFF_W22);
  unsigned short* pW23 = (unsigned short*)(ws + OFF_W23);
  float*          qterm = (float*)(ws + OFF_QT);
  unsigned short* supf  = (unsigned short*)(ws + OFF_SUP);
  unsigned short* featb = (unsigned short*)(ws + OFF_FB);

  k_pack<<<640,   256, 0, stream>>>(W11, pW11, 256, 640, 40, 163840);
  k_pack<<<800,   256, 0, stream>>>(W12, pW12, 640, 320, 20, 204800);
  k_pack<<<61504, 256, 0, stream>>>(Wg,  pWg, 3968, 3968, 248, 15745024);
  k_pack<<<4960,  256, 0, stream>>>(W21, pW21, 3968, 320, 20, 1269760);
  k_pack<<<200,   256, 0, stream>>>(W22, pW22, 320, 160, 10, 51200);
  k_pack<<<20,    256, 0, stream>>>(W23, pW23, 160, 30, 2, 5120);
  k_packf<<<1920, 256, 0, stream>>>(feat, featb, 491520);

  k_pair<<<7200, 32, 77824, stream>>>(featb, W13, b11, b12, b13, pW11, pW12, out);
  k_qterm<<<dim3(248, 4), 32, 0, stream>>>(featb, pWg, qterm);
  k_support<<<dim3(248, 64), 32, 0, stream>>>(featb, pWg, qterm, bg, supf);
  k_tail<<<dim3(4, 64), 32, 144384, stream>>>(adj, out + 115200, supf,
                                              pW21, pW22, pW23, b21, b22, b23, out);
}
